// PredictiveColumn_29961691857579
// MI455X (gfx1250) — matrix operand order repaired and checked by reading, not compiled, not run on hardware
//
#include <hip/hip_runtime.h>
#include <math.h>

#define DEV __device__ __forceinline__

static constexpr int BATCH = 16384;
static constexpr int DIM   = 512;   // DIN == DOUT == 512
static constexpr int KWTA  = 128;   // top 25% of 512

typedef __bf16 bf16x16 __attribute__((ext_vector_type(16)));
typedef float  f32x8   __attribute__((ext_vector_type(8)));

struct Bytes32 { uint4 lo, hi; };

// Assemble a 16x bf16 WMMA fragment from two 16B-aligned global loads.
DEV bf16x16 make_frag(const void* p0, const void* p1) {
  Bytes32 u;
  u.lo = *reinterpret_cast<const uint4*>(p0);
  u.hi = *reinterpret_cast<const uint4*>(p1);
  return __builtin_bit_cast(bf16x16, u);
}

DEV f32x8 wmma_bf16(bf16x16 a, bf16x16 b, f32x8 c) {
  // 8-arg form: (neg_a, A, neg_b, B, c_mod, C, reuse_a, reuse_b)
  return __builtin_amdgcn_wmma_f32_16x16x32_bf16(false, a, false, b, (short)0, c,
                                                 false, false);
}

// fp32 -> bf16(hi) + bf16(lo) split (RNE via bit trick); hi+lo ~ fp32 accuracy.
DEV void bf16_split(float x, unsigned short& h, unsigned short& l) {
  unsigned int u = __float_as_uint(x);
  unsigned int r = u + 0x7FFFu + ((u >> 16) & 1u);
  h = (unsigned short)(r >> 16);
  float hf = __uint_as_float(((unsigned int)h) << 16);
  float res = x - hf;
  unsigned int u2 = __float_as_uint(res);
  unsigned int r2 = u2 + 0x7FFFu + ((u2 >> 16) & 1u);
  l = (unsigned short)(r2 >> 16);
}

// Pack 16 u16 values into two 16B stores (dst 32B-aligned).
DEV void pack_store16(unsigned short* dst, const unsigned short* v) {
  uint4 a, b;
  a.x = (unsigned)v[0]  | ((unsigned)v[1]  << 16);
  a.y = (unsigned)v[2]  | ((unsigned)v[3]  << 16);
  a.z = (unsigned)v[4]  | ((unsigned)v[5]  << 16);
  a.w = (unsigned)v[6]  | ((unsigned)v[7]  << 16);
  b.x = (unsigned)v[8]  | ((unsigned)v[9]  << 16);
  b.y = (unsigned)v[10] | ((unsigned)v[11] << 16);
  b.z = (unsigned)v[12] | ((unsigned)v[13] << 16);
  b.w = (unsigned)v[14] | ((unsigned)v[15] << 16);
  reinterpret_cast<uint4*>(dst)[0] = a;
  reinterpret_cast<uint4*>(dst)[1] = b;
}

// ---------------------------------------------------------------------------
// K0: build bf16 hi/lo weight planes, all stored N-major (Bp[n][k], k contig)
//   pred:    Bp[i][k] = W[k][i]              (N = DIN)
//   feedback:Bp[j][k] = W[j][k]              (N = DOUT)
//   forward: Bp[j][k] = V[k][j]              (N = DOUT)
//   lateral: Bp[j][k] = (L*L_mask)[k][j]     (N = DOUT)
// ---------------------------------------------------------------------------
__global__ void prep_weights_kernel(
    const float* __restrict__ V, const float* __restrict__ W,
    const float* __restrict__ L, const float* __restrict__ Lm,
    unsigned short* __restrict__ WpH, unsigned short* __restrict__ WpL,
    unsigned short* __restrict__ WfH, unsigned short* __restrict__ WfL,
    unsigned short* __restrict__ WvH, unsigned short* __restrict__ WvL,
    unsigned short* __restrict__ WlH, unsigned short* __restrict__ WlL) {
  int t = blockIdx.x * blockDim.x + threadIdx.x;
  if (t >= DIM * DIM) return;
  int n = t >> 9, k = t & 511;
  bf16_split(W[k * DIM + n], WpH[t], WpL[t]);
  bf16_split(W[n * DIM + k], WfH[t], WfL[t]);
  bf16_split(V[k * DIM + n], WvH[t], WvL[t]);
  bf16_split(L[k * DIM + n] * Lm[k * DIM + n], WlH[t], WlL[t]);
}

// ---------------------------------------------------------------------------
// K1: one wave per row. Exact top-K threshold by 32-step bit-search on
// monotonic uint keys; ties resolved lowest-index-first (matches lax.top_k).
// Emits phi hi/lo bf16 planes, phi' fp32, and bottom_up hi/lo bf16 planes.
// ---------------------------------------------------------------------------
__global__ void mask_act_kernel(
    const float* __restrict__ x, const float* __restrict__ bu,
    unsigned short* __restrict__ phiH, unsigned short* __restrict__ phiL,
    unsigned short* __restrict__ buH,  unsigned short* __restrict__ buL,
    float* __restrict__ phiD) {
  const int lane = threadIdx.x & 31;
  const int row  = blockIdx.x * 8 + (threadIdx.x >> 5);
  const size_t rbase = (size_t)row * DIM + lane * 16;

  float xv[16];
  const float4* xp = reinterpret_cast<const float4*>(x + rbase);
#pragma unroll
  for (int i = 0; i < 4; ++i) {
    float4 v = xp[i];
    xv[4 * i + 0] = v.x; xv[4 * i + 1] = v.y;
    xv[4 * i + 2] = v.z; xv[4 * i + 3] = v.w;
  }
  unsigned key[16];
#pragma unroll
  for (int q = 0; q < 16; ++q) {
    unsigned u = __float_as_uint(xv[q]);
    key[q] = (u & 0x80000000u) ? ~u : (u | 0x80000000u);
  }
  // bit-wise binary search for the KWTA-th largest key (all lanes converge)
  unsigned thr = 0u;
  for (int bit = 31; bit >= 0; --bit) {
    unsigned cand = thr | (1u << bit);
    int c = 0;
#pragma unroll
    for (int q = 0; q < 16; ++q) c += (key[q] >= cand) ? 1 : 0;
#pragma unroll
    for (int m = 16; m; m >>= 1) c += __shfl_xor(c, m, 32);
    if (c >= KWTA) thr = cand;
  }
  // exact-K tie handling, lowest index first
  int gt = 0, eqc = 0;
#pragma unroll
  for (int q = 0; q < 16; ++q) { gt += (key[q] > thr); eqc += (key[q] == thr); }
  int gts = gt;
#pragma unroll
  for (int m = 16; m; m >>= 1) gts += __shfl_xor(gts, m, 32);
  int incl = eqc;
  for (int d = 1; d < 32; d <<= 1) {
    int t = __shfl_up(incl, d, 32);
    if (lane >= d) incl += t;
  }
  const int ties_need = KWTA - gts;
  const int eq_before = incl - eqc;

  unsigned short hh[16], hl[16];
  float dphi[16];
  int run = 0;
#pragma unroll
  for (int q = 0; q < 16; ++q) {
    const bool eq = (key[q] == thr);
    const bool in = (key[q] > thr) || (eq && (eq_before + run) < ties_need);
    run += eq ? 1 : 0;
    const float m  = in ? 1.0f : 0.0f;
    const float xq = xv[q];
    const float cdf = 0.5f * (1.0f + erff(xq * 0.70710678118654752f));
    const float pdf = expf(-0.5f * xq * xq) * 0.39894228040143268f;
    const float phi = xq * cdf * m;          // exact GELU * mask
    dphi[q] = (cdf + xq * pdf) * m;
    bf16_split(phi, hh[q], hl[q]);
  }
  pack_store16(phiH + rbase, hh);
  pack_store16(phiL + rbase, hl);
  float4* dp = reinterpret_cast<float4*>(phiD + rbase);
#pragma unroll
  for (int i = 0; i < 4; ++i) {
    float4 v;
    v.x = dphi[4 * i + 0]; v.y = dphi[4 * i + 1];
    v.z = dphi[4 * i + 2]; v.w = dphi[4 * i + 3];
    dp[i] = v;
  }
  // bottom_up -> bf16 hi/lo planes
  const float4* bp = reinterpret_cast<const float4*>(bu + rbase);
  unsigned short bh[16], bl[16];
#pragma unroll
  for (int i = 0; i < 4; ++i) {
    float4 v = bp[i];
    bf16_split(v.x, bh[4 * i + 0], bl[4 * i + 0]);
    bf16_split(v.y, bh[4 * i + 1], bl[4 * i + 1]);
    bf16_split(v.z, bh[4 * i + 2], bl[4 * i + 2]);
    bf16_split(v.w, bh[4 * i + 3], bl[4 * i + 3]);
  }
  pack_store16(buH + rbase, bh);
  pack_store16(buL + rbase, bl);
}

// ---------------------------------------------------------------------------
// K2: prediction GEMM + error.  One 16x16 output tile per wave.
// error[b][i] = bu[b][i] - (phi @ W + b_out)[b][i], stored as bf16 hi/lo.
// ---------------------------------------------------------------------------
__global__ void pred_err_kernel(
    const float* __restrict__ bu, const float* __restrict__ b_out,
    const unsigned short* __restrict__ phiH, const unsigned short* __restrict__ phiL,
    const unsigned short* __restrict__ WpH,  const unsigned short* __restrict__ WpL,
    unsigned short* __restrict__ errH, unsigned short* __restrict__ errL) {
  const int lane  = threadIdx.x & 31;
  const int wid   = (blockIdx.x * blockDim.x + threadIdx.x) >> 5;
  const int tileM = wid >> 5;   // 1024 row tiles
  const int tileN = wid & 31;   // 32 col tiles (DIN)
  const int hsel  = lane >> 4;
  const int lm    = lane & 15;

  const char* aH = (const char*)phiH + (size_t)(tileM * 16 + lm) * 1024;
  const char* aL = (const char*)phiL + (size_t)(tileM * 16 + lm) * 1024;
  const char* bH = (const char*)WpH  + (size_t)(tileN * 16 + lm) * 1024;
  const char* bL = (const char*)WpL  + (size_t)(tileN * 16 + lm) * 1024;

  f32x8 acc = {};
  for (int kc = 0; kc < 16; ++kc) {
    const int ao = kc * 64 + hsel * 32;
    const int bo = kc * 64 + hsel * 32;
    bf16x16 fAH = make_frag(aH + ao, aH + ao + 16);
    bf16x16 fAL = make_frag(aL + ao, aL + ao + 16);
    bf16x16 fBH = make_frag(bH + bo, bH + bo + 16);
    bf16x16 fBL = make_frag(bL + bo, bL + bo + 16);
    acc = wmma_bf16(fAH, fBH, acc);
    acc = wmma_bf16(fAH, fBL, acc);
    acc = wmma_bf16(fAL, fBH, acc);
  }
  const int n = tileN * 16 + lm;
  const float bo_v = b_out[n];
#pragma unroll
  for (int r = 0; r < 8; ++r) {
    const int brow = tileM * 16 + r + 8 * hsel;   // D layout: M = r + 8*half
    const size_t idx = (size_t)brow * DIM + n;    // N = lane%16
    const float err = bu[idx] - (acc[r] + bo_v);
    bf16_split(err, errH[idx], errL[idx]);
  }
}

// ---------------------------------------------------------------------------
// K3: fused 3-GEMM drives + combine.  One 16x16 output tile per wave.
//   accF = err @ W^T ; accV = bu @ V ; accL = phi @ (L*mask)
//   g = accF*phi' + (accV + b_in - x) + 3(td - x) + 0.3 accL - 1e-6 sign(x)
//   dx_pre = tau*eta*g ; per-block deterministic partial sum of dx_pre^2.
// ---------------------------------------------------------------------------
__global__ void drives_kernel(
    const float* __restrict__ x, const float* __restrict__ tdt,
    const float* __restrict__ b_in, const float* __restrict__ phiD,
    const int* __restrict__ step_i,
    const unsigned short* __restrict__ errH, const unsigned short* __restrict__ errL,
    const unsigned short* __restrict__ buH,  const unsigned short* __restrict__ buL,
    const unsigned short* __restrict__ phiH, const unsigned short* __restrict__ phiL,
    const unsigned short* __restrict__ WfH,  const unsigned short* __restrict__ WfL,
    const unsigned short* __restrict__ WvH,  const unsigned short* __restrict__ WvL,
    const unsigned short* __restrict__ WlH,  const unsigned short* __restrict__ WlL,
    float* __restrict__ dxp, float* __restrict__ partials) {
  const int lane  = threadIdx.x & 31;
  const int wid   = (blockIdx.x * blockDim.x + threadIdx.x) >> 5;
  const int tileM = wid >> 5;
  const int tileN = wid & 31;
  const int hsel  = lane >> 4;
  const int lm    = lane & 15;

  const size_t arow = (size_t)(tileM * 16 + lm) * 1024;
  const size_t brow = (size_t)(tileN * 16 + lm) * 1024;
  const char* e0 = (const char*)errH + arow; const char* e1 = (const char*)errL + arow;
  const char* u0 = (const char*)buH  + arow; const char* u1 = (const char*)buL  + arow;
  const char* p0 = (const char*)phiH + arow; const char* p1 = (const char*)phiL + arow;
  const char* f0 = (const char*)WfH  + brow; const char* f1 = (const char*)WfL  + brow;
  const char* v0 = (const char*)WvH  + brow; const char* v1 = (const char*)WvL  + brow;
  const char* l0 = (const char*)WlH  + brow; const char* l1 = (const char*)WlL  + brow;

  f32x8 accF = {}, accV = {}, accLt = {};
  for (int kc = 0; kc < 16; ++kc) {
    const int ao = kc * 64 + hsel * 32;
    const int bo = kc * 64 + hsel * 32;
    bf16x16 EH = make_frag(e0 + ao, e0 + ao + 16);
    bf16x16 EL = make_frag(e1 + ao, e1 + ao + 16);
    bf16x16 UH = make_frag(u0 + ao, u0 + ao + 16);
    bf16x16 UL = make_frag(u1 + ao, u1 + ao + 16);
    bf16x16 PH = make_frag(p0 + ao, p0 + ao + 16);
    bf16x16 PL = make_frag(p1 + ao, p1 + ao + 16);
    bf16x16 FH = make_frag(f0 + bo, f0 + bo + 16);
    bf16x16 FL = make_frag(f1 + bo, f1 + bo + 16);
    bf16x16 VH = make_frag(v0 + bo, v0 + bo + 16);
    bf16x16 VL = make_frag(v1 + bo, v1 + bo + 16);
    bf16x16 LH = make_frag(l0 + bo, l0 + bo + 16);
    bf16x16 LL = make_frag(l1 + bo, l1 + bo + 16);
    accF  = wmma_bf16(EH, FH, accF);  accF  = wmma_bf16(EH, FL, accF);  accF  = wmma_bf16(EL, FH, accF);
    accV  = wmma_bf16(UH, VH, accV);  accV  = wmma_bf16(UH, VL, accV);  accV  = wmma_bf16(UL, VH, accV);
    accLt = wmma_bf16(PH, LH, accLt); accLt = wmma_bf16(PH, LL, accLt); accLt = wmma_bf16(PL, LH, accLt);
  }

  const float eta = 0.8f / (1.0f + 0.1f * (float)step_i[0]);
  const float s   = 0.5f * eta;           // tau * eta
  const int n = tileN * 16 + lm;
  const float bi = b_in[n];
  float ss = 0.0f;
#pragma unroll
  for (int r = 0; r < 8; ++r) {
    const int brow_m = tileM * 16 + r + 8 * hsel;
    const size_t idx = (size_t)brow_m * DIM + n;
    const float xv  = x[idx];
    const float sgn = (xv > 0.0f ? 1.0f : 0.0f) - (xv < 0.0f ? 1.0f : 0.0f);
    const float g = accF[r] * phiD[idx]
                  + (accV[r] + bi - xv)
                  + 3.0f * (tdt[idx] - xv)
                  + 0.3f * accLt[r]
                  - 1e-6f * sgn;
    const float d = s * g;
    dxp[idx] = d;
    ss += d * d;
  }
#pragma unroll
  for (int m = 16; m; m >>= 1) ss += __shfl_xor(ss, m, 32);
  __shared__ float sm[8];
  if (lane == 0) sm[threadIdx.x >> 5] = ss;
  __syncthreads();
  if (threadIdx.x == 0) {
    float t = 0.0f;
#pragma unroll
    for (int i = 0; i < 8; ++i) t += sm[i];
    partials[blockIdx.x] = t;
  }
}

// K3b: deterministic single-block tree reduction of 4096 partials.
__global__ void reduce_partials_kernel(const float* __restrict__ partials,
                                       float* __restrict__ sumsq) {
  __shared__ float sm[256];
  float v = 0.0f;
  for (int i = threadIdx.x; i < 4096; i += 256) v += partials[i];
  sm[threadIdx.x] = v;
  __syncthreads();
  for (int st = 128; st > 0; st >>= 1) {
    if ((int)threadIdx.x < st) sm[threadIdx.x] += sm[threadIdx.x + st];
    __syncthreads();
  }
  if (threadIdx.x == 0) sumsq[0] = sm[0];
}

// K4: global-norm clip + clamp.
__global__ void finalize_kernel(const float* __restrict__ x,
                                const float* __restrict__ dxp,
                                const float* __restrict__ sumsq,
                                float* __restrict__ out, int nTot) {
  int i = blockIdx.x * blockDim.x + threadIdx.x;
  if (i >= nTot) return;
  const float nrm = sqrtf(sumsq[0]);
  const float sc  = (nrm > 1.0f) ? (1.0f / (nrm + 1e-8f)) : 1.0f;
  const float v = x[i] + dxp[i] * sc;
  out[i] = fminf(fmaxf(v, -5.0f), 5.0f);
}

// ---------------------------------------------------------------------------
extern "C" void kernel_launch(void* const* d_in, const int* in_sizes, int n_in,
                              void* d_out, int out_size, void* d_ws, size_t ws_size,
                              hipStream_t stream) {
  (void)in_sizes; (void)n_in; (void)out_size; (void)ws_size;
  const float* bu    = (const float*)d_in[0];
  const float* tdt   = (const float*)d_in[1];
  const float* x     = (const float*)d_in[2];
  const float* V     = (const float*)d_in[3];
  const float* b_in  = (const float*)d_in[4];
  const float* W     = (const float*)d_in[5];
  const float* b_out = (const float*)d_in[6];
  const float* L     = (const float*)d_in[7];
  const float* Lm    = (const float*)d_in[8];
  const int*   step  = (const int*)d_in[9];

  char* ws = (char*)d_ws;
  float* sumsq    = (float*)ws;             // 1 float
  float* partials = (float*)(ws + 256);     // 4096 floats
  size_t off = 256 + 4096 * 4;              // 256-aligned
  const size_t WPLANE = (size_t)DIM * DIM * 2;
  unsigned short* WpH = (unsigned short*)(ws + off); off += WPLANE;
  unsigned short* WpL = (unsigned short*)(ws + off); off += WPLANE;
  unsigned short* WfH = (unsigned short*)(ws + off); off += WPLANE;
  unsigned short* WfL = (unsigned short*)(ws + off); off += WPLANE;
  unsigned short* WvH = (unsigned short*)(ws + off); off += WPLANE;
  unsigned short* WvL = (unsigned short*)(ws + off); off += WPLANE;
  unsigned short* WlH = (unsigned short*)(ws + off); off += WPLANE;
  unsigned short* WlL = (unsigned short*)(ws + off); off += WPLANE;
  const size_t P16 = (size_t)BATCH * DIM * 2;
  unsigned short* phiH = (unsigned short*)(ws + off); off += P16;
  unsigned short* phiL = (unsigned short*)(ws + off); off += P16;
  unsigned short* buH  = (unsigned short*)(ws + off); off += P16;
  unsigned short* buL  = (unsigned short*)(ws + off); off += P16;
  unsigned short* errH = (unsigned short*)(ws + off); off += P16;
  unsigned short* errL = (unsigned short*)(ws + off); off += P16;
  const size_t P32 = (size_t)BATCH * DIM * 4;
  float* phiD = (float*)(ws + off); off += P32;
  float* dxp  = (float*)(ws + off); off += P32;

  prep_weights_kernel<<<1024, 256, 0, stream>>>(V, W, L, Lm, WpH, WpL, WfH, WfL,
                                                WvH, WvL, WlH, WlL);
  mask_act_kernel<<<2048, 256, 0, stream>>>(x, bu, phiH, phiL, buH, buL, phiD);
  pred_err_kernel<<<4096, 256, 0, stream>>>(bu, b_out, phiH, phiL, WpH, WpL,
                                            errH, errL);
  drives_kernel<<<4096, 256, 0, stream>>>(x, tdt, b_in, phiD, step,
                                          errH, errL, buH, buL, phiH, phiL,
                                          WfH, WfL, WvH, WvL, WlH, WlL,
                                          dxp, partials);
  reduce_partials_kernel<<<1, 256, 0, stream>>>(partials, sumsq);
  finalize_kernel<<<32768, 256, 0, stream>>>(x, dxp, sumsq, (float*)d_out,
                                             BATCH * DIM);
}
